// TGCN_60962765799966
// MI455X (gfx1250) — matrix operand order repaired and checked by reading, not compiled, not run on hardware
//
#include <hip/hip_runtime.h>

// ---------------------------------------------------------------------------
// TGCN forward for MI455X (gfx1250, wave32).
//   Stage 1: GCN (deg/norm, x@W1 dot, 2x edge-aggregation w/ native f32
//            atomics into L2-resident buffers, sigmoid, BatchNorm) -> bf16.
//   Stage 2: big WMMA bf16 GEMMs for GRU input projections (GI = o@Wih^T+bih).
//   Stage 3: persistent-grid GRU recurrence: 16 WGs, device barrier per step,
//            3x v_wmma_f32_16x16x32_bf16 per K-chunk, gate fusion in regs.
//   Stage 4: final Linear(H->1) with wave reduction.
// Roofline: x read (256 MB -> ~11 us @ 23.3 TB/s) is the HBM floor; all other
// hot buffers (16 MB node matrices, 25 MB weights, 12.6 MB GI) live in the
// 192 MB L2. GEMM FLOPs (~23 G) are negligible in bf16 WMMA; the sequential
// GRU path runs as one persistent kernel to avoid 64 launch round-trips.
// Workspace use: ~59 MB.
// ---------------------------------------------------------------------------

constexpr int S  = 32;
constexpr int B  = 64;
constexpr int N  = 2048;
constexpr int F  = 16;
constexpr int E  = 32768;
constexpr int H  = 512;
constexpr int SB = S * B;        // 2048
constexpr int G3 = 3 * H;        // 1536
constexpr int EN = E + N;        // 34816 (edges + self loops)
constexpr int GRU_WGS = 16;      // persistent workgroups for the recurrence

typedef __attribute__((ext_vector_type(16))) __bf16 v16bf;
typedef __attribute__((ext_vector_type(8)))  __bf16 v8bf;
typedef __attribute__((ext_vector_type(8)))  float  v8f;

// ------------------------------- WMMA helpers ------------------------------

__device__ __forceinline__ v8f wmma_bf16(v16bf a, v16bf b, v8f c) {
  // D = A(16x32) * B(32x16) + C(16x16 f32). 8-arg probe-confirmed signature.
  return __builtin_amdgcn_wmma_f32_16x16x32_bf16(
      /*neg_a=*/false, a, /*neg_b=*/false, b,
      /*c_mod=*/(short)0, c, /*reuse_a=*/false, /*reuse_b=*/false);
}

// A fragment: row-major A[M][K] bf16 tile at (m0,k0). Contiguous k order (as load_b):
// lanes 0-15: M=lane, K = k0..k0+15; lanes 16-31: M=lane-16, K = k0+16..k0+31.
__device__ __forceinline__ v16bf load_a(const __bf16* __restrict__ A, int ldk,
                                        int m0, int k0, int lane) {
  int lo = lane & 15, hs = lane >> 4;
  const __bf16* p = A + (size_t)(m0 + lo) * ldk + k0 + hs * 16;
  v8bf x0 = *(const v8bf*)(p);        // 16B load
  v8bf x1 = *(const v8bf*)(p + 8);    // 16B load
  v16bf r;
#pragma unroll
  for (int i = 0; i < 8; ++i) { r[i] = x0[i]; r[8 + i] = x1[i]; }
  return r;
}

// B fragment: B[k][n] = W[n][k] with W row-major [Ncols][K] bf16 (PyTorch
// weight layout -> K-contiguous per output column, one 32B load per lane).
// lanes 0-15: N=lane, K=k0..k0+15; lanes 16-31: N=lane-16, K=k0+16..k0+31.
__device__ __forceinline__ v16bf load_b(const __bf16* __restrict__ W, int ldk,
                                        int n0, int k0, int lane) {
  int lo = lane & 15, hs = lane >> 4;
  return *(const v16bf*)(W + (size_t)(n0 + lo) * ldk + k0 + hs * 16);
}

__device__ __forceinline__ float sigf(float x) {
  return 1.0f / (1.0f + __expf(-x));
}

// ------------------------------- small kernels -----------------------------

__global__ void fill_kernel(float* p, float v, int n) {
  int i = blockIdx.x * 256 + threadIdx.x;
  if (i < n) p[i] = v;
}

__global__ void cvt_bf16_kernel(const float* __restrict__ src,
                                __bf16* __restrict__ dst, int n) {
  int i = blockIdx.x * 256 + threadIdx.x;
  if (i < n) dst[i] = (__bf16)src[i];
}

__global__ void deg_count_kernel(const int* __restrict__ ei, float* deg) {
  int e = blockIdx.x * 256 + threadIdx.x;
  if (e < E) unsafeAtomicAdd(&deg[ei[E + e]], 1.0f);   // native f32 atomic
}

__global__ void dinv_kernel(float* deg) {
  int n = blockIdx.x * 256 + threadIdx.x;
  if (n < N) deg[n] = rsqrtf(deg[n]);
}

__global__ void norm_kernel(const int* __restrict__ ei,
                            const float* __restrict__ dinv, float* nrm) {
  int e = blockIdx.x * 256 + threadIdx.x;
  if (e >= EN) return;
  int s, d;
  if (e < E) { s = ei[e]; d = ei[E + e]; } else { s = d = e - E; }
  nrm[e] = dinv[s] * dinv[d];
}

// xw[n][sb] = sum_f x[sb][n][f] * W1[f]   (GH == 1)
// Per-lane 64B contiguous read of x (exactly-once HBM traffic), coalesced write.
__global__ __launch_bounds__(256) void xw_kernel(const float* __restrict__ x,
                                                 const float* __restrict__ W1,
                                                 float* __restrict__ xw) {
  int n  = blockIdx.x;                       // 0..N-1
  int sb = blockIdx.y * 256 + threadIdx.x;   // 0..SB-1
  const float4* xp = (const float4*)(x + ((size_t)sb * N + n) * F);
  float4 v0 = xp[0], v1 = xp[1], v2 = xp[2], v3 = xp[3];
  float s = v0.x * W1[0] + v0.y * W1[1] + v0.z * W1[2] + v0.w * W1[3]
          + v1.x * W1[4] + v1.y * W1[5] + v1.z * W1[6] + v1.w * W1[7]
          + v2.x * W1[8] + v2.y * W1[9] + v2.z * W1[10] + v2.w * W1[11]
          + v3.x * W1[12] + v3.y * W1[13] + v3.z * W1[14] + v3.w * W1[15];
  xw[(size_t)n * SB + sb] = s;
}

// out[dst][sb] += norm[e]*scale * in[src][sb]; one block per edge, both
// matrices (16 MB) are L2-resident; native global_atomic_add_f32 scatter.
__global__ __launch_bounds__(256) void agg_kernel(const int* __restrict__ ei,
                                                  const float* __restrict__ nrm,
                                                  const float* __restrict__ hin,
                                                  float* __restrict__ hout,
                                                  const float* scale_ptr) {
  int e = blockIdx.x;
  int s, d;
  if (e < E) { s = ei[e]; d = ei[E + e]; } else { s = d = e - E; }
  float w = nrm[e] * (scale_ptr ? scale_ptr[0] : 1.0f);
  const float* in = hin + (size_t)s * SB;
  float* outp = hout + (size_t)d * SB;
  for (int i = threadIdx.x; i < SB; i += 256)
    unsafeAtomicAdd(outp + i, w * in[i]);
}

__global__ void sigb_kernel(float* p, const float* b1, int n) {
  int i = blockIdx.x * 256 + threadIdx.x;
  if (i < n) p[i] = sigf(p[i] + b1[0]);
}

// BatchNorm (eval) + transpose (n,sb)->(sb,n) + bf16 cast for the GI0 GEMM.
__global__ __launch_bounds__(256) void bn_kernel(const float* __restrict__ agg2,
                                                 const float* b2,
                                                 const float* __restrict__ gamma,
                                                 const float* __restrict__ beta,
                                                 const float* __restrict__ rmean,
                                                 const float* __restrict__ rvar,
                                                 __bf16* __restrict__ obf) {
  int sb = blockIdx.x;
  int n  = blockIdx.y * 256 + threadIdx.x;
  float v = agg2[(size_t)n * SB + sb] + b2[0];
  v = (v - rmean[n]) * rsqrtf(rvar[n] + 1e-5f) * gamma[n] + beta[n];
  obf[(size_t)sb * N + n] = (__bf16)v;
}

// ------------------------- big WMMA GEMM (GI = A@W^T+b) --------------------
// A: (M,K) bf16 row-major. W: (Nc,K) bf16 row-major. C: (M,Nc) f32.
// 256 threads = 8 waves as 2(M)x4(N); wave tile 64x32 = 4x2 C-frags.
__global__ __launch_bounds__(256)
void wmma_gemm_kernel(const __bf16* __restrict__ A, const __bf16* __restrict__ W,
                      const float* __restrict__ bias, float* __restrict__ C,
                      int M, int Nc, int K) {
  int tid = threadIdx.x;
  int lane = tid & 31, wv = tid >> 5;
  int wm = wv & 1, wn = wv >> 1;
  int m0 = blockIdx.x * 128 + wm * 64;
  int n0 = blockIdx.y * 128 + wn * 32;
  v8f acc[4][2] = {};
  for (int k0 = 0; k0 < K; k0 += 32) {
    v16bf a[4], b[2];
#pragma unroll
    for (int i = 0; i < 4; ++i) a[i] = load_a(A, K, m0 + i * 16, k0, lane);
#pragma unroll
    for (int j = 0; j < 2; ++j) b[j] = load_b(W, K, n0 + j * 16, k0, lane);
#pragma unroll
    for (int i = 0; i < 4; ++i)
#pragma unroll
      for (int j = 0; j < 2; ++j) acc[i][j] = wmma_bf16(a[i], b[j], acc[i][j]);
  }
  int lo = lane & 15, hs = lane >> 4;
#pragma unroll
  for (int i = 0; i < 4; ++i)
#pragma unroll
    for (int j = 0; j < 2; ++j) {
      int col = n0 + j * 16 + lo;
      float bv = bias ? bias[col] : 0.0f;
#pragma unroll
      for (int v = 0; v < 8; ++v) {
        int row = m0 + i * 16 + hs * 8 + v;
        C[(size_t)row * Nc + col] = acc[i][j][v] + bv;
      }
    }
}

// ------------------------------ GRU recurrence -----------------------------

__global__ void gru_init_kernel(float* hf0, __bf16* hb0, unsigned* bar) {
  int i = blockIdx.x * 256 + threadIdx.x;
  if (i < B * H) { hf0[i] = 0.0f; hb0[i] = (__bf16)0.0f; }
  if (i == 0) *bar = 0u;
}

// Persistent kernel: GRU_WGS workgroups, each owns a 32-col slice of H.
// Per step: gh tiles via 3 WMMA chains (r,z,n gates, K=512), gate fusion on
// accumulator registers, ping-pong h buffers, device barrier between steps.
__global__ __launch_bounds__(256)
void gru_seq_kernel(const float* __restrict__ GI,   // (S,B,1536) f32
                    const __bf16* __restrict__ Whh, // (1536,512) bf16
                    const float* __restrict__ bhh,  // (1536,)
                    float* hf0, float* hf1, __bf16* hb0, __bf16* hb1,
                    __bf16* __restrict__ ybf,       // (S,B,512) bf16
                    float* __restrict__ yf,         // (S,B,512) f32
                    float* __restrict__ hn_out,     // (B,512) f32 (d_out slice)
                    unsigned* bar) {
  const int tid = threadIdx.x;
  const int lane = tid & 31, wv = tid >> 5;
  const int mi = wv & 3, nj = wv >> 2;         // 4(M) x 2(N) waves
  const int lo = lane & 15, hs = lane >> 4;
  const int c0 = blockIdx.x * 32;              // this WG's H-slice
  const int m0 = mi * 16;
  const int jb = c0 + nj * 16;
  const int j  = jb + lo;                      // output column in [0,H)
  const float bhr = bhh[j], bhz = bhh[H + j], bhn = bhh[2 * H + j];

  for (int t = 0; t < S; ++t) {
    const __bf16* hbr = (t & 1) ? hb1 : hb0;   // read buffer (bf16, GEMM A)
    const float*  hfr = (t & 1) ? hf1 : hf0;   // read buffer (f32, hprev)
    __bf16* hbw = (t & 1) ? hb0 : hb1;
    float*  hfw = (t & 1) ? hf0 : hf1;

    v8f cr = {}, cz = {}, cn = {};
    for (int k0 = 0; k0 < H; k0 += 32) {
      v16bf a  = load_a(hbr, H, m0, k0, lane);
      v16bf br = load_b(Whh,                  H, jb, k0, lane);
      v16bf bz = load_b(Whh + (size_t)H * H,  H, jb, k0, lane);
      v16bf bn = load_b(Whh + (size_t)2*H*H,  H, jb, k0, lane);
      cr = wmma_bf16(a, br, cr);
      cz = wmma_bf16(a, bz, cz);
      cn = wmma_bf16(a, bn, cn);
    }

    const float* GIt = GI + (size_t)t * B * G3;
    const bool last = (t == S - 1);
#pragma unroll
    for (int v = 0; v < 8; ++v) {
      int brow = m0 + hs * 8 + v;                  // batch row
      const float* gi = GIt + (size_t)brow * G3;   // includes bih
      float r  = sigf(gi[j] + cr[v] + bhr);
      float z  = sigf(gi[H + j] + cz[v] + bhz);
      float hn = cn[v] + bhn;
      float nn = tanhf(gi[2 * H + j] + r * hn);
      float hp = hfr[(size_t)brow * H + j];
      float hnew = (1.0f - z) * nn + z * hp;
      hfw[(size_t)brow * H + j] = hnew;
      hbw[(size_t)brow * H + j] = (__bf16)hnew;
      size_t yo = ((size_t)t * B + brow) * H + j;
      yf[yo] = hnew;
      ybf[yo] = (__bf16)hnew;
      if (last) hn_out[(size_t)brow * H + j] = hnew;
    }

    // device-scope generation barrier: all h writes visible before next step
    __threadfence();
    __syncthreads();
    if (tid == 0) {
      atomicAdd(bar, 1u);
      unsigned target = (unsigned)(t + 1) * gridDim.x;
      while (__hip_atomic_load(bar, __ATOMIC_ACQUIRE,
                               __HIP_MEMORY_SCOPE_AGENT) < target)
        __builtin_amdgcn_s_sleep(1);
    }
    __syncthreads();
    __threadfence();
  }
}

// out[sb] = dot(y2[sb], Wlin) + blin ; one wave per row, shfl reduction.
__global__ __launch_bounds__(256)
void linear_kernel(const float* __restrict__ y, const float* __restrict__ Wlin,
                   const float* blin, float* __restrict__ out) {
  int lane = threadIdx.x & 31, wv = threadIdx.x >> 5;
  int sb = blockIdx.x * 8 + wv;
  const float* row = y + (size_t)sb * H;
  float s = 0.0f;
  for (int k = lane; k < H; k += 32) s += row[k] * Wlin[k];
#pragma unroll
  for (int off = 16; off; off >>= 1) s += __shfl_xor(s, off, 32);
  if (lane == 0) out[sb] = s + blin[0];
}

// --------------------------------- launcher --------------------------------

extern "C" void kernel_launch(void* const* d_in, const int* in_sizes, int n_in,
                              void* d_out, int out_size, void* d_ws, size_t ws_size,
                              hipStream_t stream) {
  (void)in_sizes; (void)n_in; (void)out_size; (void)ws_size;
  const float* x     = (const float*)d_in[0];
  const int*   ei    = (const int*)d_in[1];   // (2,E): row0=src, row1=dst
  const float* W1    = (const float*)d_in[2];
  const float* b1    = (const float*)d_in[3];
  const float* W2    = (const float*)d_in[4];
  const float* b2    = (const float*)d_in[5];
  const float* gamma = (const float*)d_in[6];
  const float* beta  = (const float*)d_in[7];
  const float* rmean = (const float*)d_in[8];
  const float* rvar  = (const float*)d_in[9];
  const float* Wih0  = (const float*)d_in[10];
  const float* Whh0  = (const float*)d_in[11];
  const float* bih0  = (const float*)d_in[12];
  const float* bhh0  = (const float*)d_in[13];
  const float* Wih1  = (const float*)d_in[14];
  const float* Whh1  = (const float*)d_in[15];
  const float* bih1  = (const float*)d_in[16];
  const float* bhh1  = (const float*)d_in[17];
  const float* Wlin  = (const float*)d_in[18];
  const float* blin  = (const float*)d_in[19];
  float* out = (float*)d_out;

  char* ws = (char*)d_ws;
  size_t off = 0;
  auto alloc = [&](size_t bytes) {
    size_t r = off;
    off += (bytes + 255) & ~(size_t)255;
    return r;
  };
  float*   deg   = (float*)  (ws + alloc((size_t)N * 4));
  float*   nrm   = (float*)  (ws + alloc((size_t)EN * 4));
  float*   bufA  = (float*)  (ws + alloc((size_t)N * SB * 4));  // xw / agg2 / GI0
  float*   bufB  = (float*)  (ws + alloc((size_t)N * SB * 4));  // agg1,h1 / GI1
  __bf16*  obf   = (__bf16*) (ws + alloc((size_t)SB * N * 2));
  __bf16*  Wih0b = (__bf16*) (ws + alloc((size_t)G3 * N * 2));
  __bf16*  Whh0b = (__bf16*) (ws + alloc((size_t)G3 * H * 2));
  __bf16*  Wih1b = (__bf16*) (ws + alloc((size_t)G3 * H * 2));
  __bf16*  Whh1b = (__bf16*) (ws + alloc((size_t)G3 * H * 2));
  __bf16*  ybf   = (__bf16*) (ws + alloc((size_t)SB * H * 2));
  float*   yf    = (float*)  (ws + alloc((size_t)SB * H * 4));
  float*   hf0   = (float*)  (ws + alloc((size_t)B * H * 4));
  float*   hf1   = (float*)  (ws + alloc((size_t)B * H * 4));
  __bf16*  hb0   = (__bf16*) (ws + alloc((size_t)B * H * 2));
  __bf16*  hb1   = (__bf16*) (ws + alloc((size_t)B * H * 2));
  unsigned* bar0 = (unsigned*)(ws + alloc(256));
  unsigned* bar1 = (unsigned*)(ws + alloc(256));
  float* GI0 = bufA;   // overlay: agg buffers dead after BatchNorm
  float* GI1 = bufB;

  dim3 blk(256);
  const int NSB = N * SB;  // 4,194,304

  // ---- GCN normalization coefficients
  fill_kernel<<<(N + 255) / 256, blk, 0, stream>>>(deg, 1.0f, N);  // self loops
  deg_count_kernel<<<(E + 255) / 256, blk, 0, stream>>>(ei, deg);
  dinv_kernel<<<(N + 255) / 256, blk, 0, stream>>>(deg);
  norm_kernel<<<(EN + 255) / 256, blk, 0, stream>>>(ei, deg, nrm);

  // ---- GCN layer 1: xw -> aggregate -> sigmoid(+b1)
  xw_kernel<<<dim3(N, SB / 256), blk, 0, stream>>>(x, W1, bufA);
  fill_kernel<<<(NSB + 255) / 256, blk, 0, stream>>>(bufB, 0.0f, NSB);
  agg_kernel<<<EN, blk, 0, stream>>>(ei, nrm, bufA, bufB, nullptr);
  sigb_kernel<<<(NSB + 255) / 256, blk, 0, stream>>>(bufB, b1, NSB);

  // ---- GCN layer 2: (W2 is 1x1 -> fold scalar into aggregation), +b2 in BN
  fill_kernel<<<(NSB + 255) / 256, blk, 0, stream>>>(bufA, 0.0f, NSB);
  agg_kernel<<<EN, blk, 0, stream>>>(ei, nrm, bufB, bufA, W2);
  bn_kernel<<<dim3(SB, N / 256), blk, 0, stream>>>(bufA, b2, gamma, beta,
                                                   rmean, rvar, obf);

  // ---- weights to bf16 (once per launch)
  cvt_bf16_kernel<<<(G3 * N + 255) / 256, blk, 0, stream>>>(Wih0, Wih0b, G3 * N);
  cvt_bf16_kernel<<<(G3 * H + 255) / 256, blk, 0, stream>>>(Whh0, Whh0b, G3 * H);
  cvt_bf16_kernel<<<(G3 * H + 255) / 256, blk, 0, stream>>>(Wih1, Wih1b, G3 * H);
  cvt_bf16_kernel<<<(G3 * H + 255) / 256, blk, 0, stream>>>(Whh1, Whh1b, G3 * H);

  // ---- GRU layer 0
  wmma_gemm_kernel<<<dim3(SB / 128, G3 / 128), blk, 0, stream>>>(
      obf, Wih0b, bih0, GI0, SB, G3, N);
  gru_init_kernel<<<(B * H + 255) / 256, blk, 0, stream>>>(hf0, hb0, bar0);
  gru_seq_kernel<<<GRU_WGS, blk, 0, stream>>>(
      GI0, Whh0b, bhh0, hf0, hf1, hb0, hb1, ybf, yf, out + SB, bar0);

  // ---- GRU layer 1
  wmma_gemm_kernel<<<dim3(SB / 128, G3 / 128), blk, 0, stream>>>(
      ybf, Wih1b, bih1, GI1, SB, G3, H);
  gru_init_kernel<<<(B * H + 255) / 256, blk, 0, stream>>>(hf0, hb0, bar1);
  gru_seq_kernel<<<GRU_WGS, blk, 0, stream>>>(
      GI1, Whh1b, bhh1, hf0, hf1, hb0, hb1, ybf, yf, out + SB + B * H, bar1);

  // ---- final linear
  linear_kernel<<<SB / 8, blk, 0, stream>>>(yf, Wlin, blin, out);
}
